// refiner_30219389895258
// MI455X (gfx1250) — matrix operand order repaired and checked by reading, not compiled, not run on hardware
//
#include <hip/hip_runtime.h>
#include <stdint.h>

// ---------------- problem constants ----------------
#define BB   4
#define NN   4096
#define CC   128
#define K1C  16
#define K2C  8
#define TDC  64
#define CINC 262
#define CPAD 288      // 9 * 32 (K padded for 16x16x32 bf16 WMMA)
#define KSTEPS 9

typedef __attribute__((ext_vector_type(16))) __bf16 v16bf;
typedef __attribute__((ext_vector_type(8)))  float  v8f;
typedef __attribute__((ext_vector_type(4))) unsigned int v4u;
typedef __attribute__((ext_vector_type(8))) int v8i;
typedef __attribute__((ext_vector_type(4))) int v4i;

union Frag16 {
  v16bf v;
  uint4 q[2];
  unsigned short s[16];
};

__device__ __forceinline__ unsigned short f2bf(float x) {
  unsigned u = __float_as_uint(x);
  unsigned r = u + 0x7FFFu + ((u >> 16) & 1u);   // round-to-nearest-even
  return (unsigned short)(r >> 16);
}

// ---------------- kernel 1: weight conversion to padded bf16 ----------------
__global__ void __launch_bounds__(256) prep_weights(
                             const float* __restrict__ wq, const float* __restrict__ wk,
                             const float* __restrict__ wv, const float* __restrict__ w1,
                             unsigned short* __restrict__ wqb, unsigned short* __restrict__ wkb,
                             unsigned short* __restrict__ wvb, unsigned short* __restrict__ w1b) {
  int i = blockIdx.x * blockDim.x + threadIdx.x;
  const int NW = TDC * CPAD;     // 18432
  if (i < NW) {
    int row = i / CPAD, col = i % CPAD;
    wqb[i] = f2bf(col < CINC ? wq[row * CINC + col] : 0.f);
  } else if (i < 2 * NW) {
    int t = i - NW; int row = t / CPAD, col = t % CPAD;
    wkb[t] = f2bf(col < CINC ? wk[row * CINC + col] : 0.f);
  } else if (i < 3 * NW) {
    int t = i - 2 * NW; int row = t / CPAD, col = t % CPAD;
    wvb[t] = f2bf(col < CINC ? wv[row * CINC + col] : 0.f);
  } else if (i < 3 * NW + CC * TDC) {
    int t = i - 3 * NW;
    w1b[t] = f2bf(w1[t]);       // row-major [128][64]
  }
}

// ---------------- kernel 2: KNN (17 nearest incl. self, drop self) ----------------
__global__ void __launch_bounds__(256, 1) knn_kernel(
                           const float* __restrict__ xyz, int* __restrict__ nidx) {
  __shared__ float sx[NN], sy[NN], sz[NN];     // 48 KB: whole batch xyz in LDS
  const int blocksPerBatch = NN / 256;
  int b = blockIdx.x / blocksPerBatch;
  int f = (blockIdx.x % blocksPerBatch) * 256 + threadIdx.x;
  const float* xb = xyz + (size_t)b * 3 * NN;
  for (int i = threadIdx.x; i < NN; i += 256) {
    sx[i] = xb[i]; sy[i] = xb[NN + i]; sz[i] = xb[2 * NN + i];
  }
  __syncthreads();
  float qx = sx[f], qy = sy[f], qz = sz[f];
  float bd[K1C + 1]; int bi[K1C + 1];
#pragma unroll
  for (int t = 0; t <= K1C; ++t) { bd[t] = 3.4e38f; bi[t] = -1; }
  for (int m = 0; m < NN; ++m) {
    float dx = sx[m] - qx, dy = sy[m] - qy, dz = sz[m] - qz;
    float d = dx * dx + dy * dy + dz * dz;
    if (d < bd[K1C]) {          // strict '<' keeps earlier index on ties (top_k stability)
      int pos = K1C;
#pragma unroll
      for (int t = K1C - 1; t >= 0; --t) if (d < bd[t]) pos = t;
#pragma unroll
      for (int t = K1C; t > 0; --t) if (t > pos) { bd[t] = bd[t - 1]; bi[t] = bi[t - 1]; }
      bd[pos] = d; bi[pos] = m;
    }
  }
  int* op = nidx + ((size_t)b * NN + f) * K1C;
#pragma unroll
  for (int t = 1; t <= K1C; ++t) op[t - 1] = bi[t];   // drop self (nearest)
}

// ---------------- kernel 3: fused gather + qkv WMMA + attention -> res ----------------
// One wave per point; 4 waves per block. Weights staged to LDS once per block via the
// Tensor Data Mover (TENSORcnt path), then A-fragments come from ds_load_b128.
// __launch_bounds__(128, 1): release the full 256-VGPR file so the 96 accumulator
// VGPRs (q/k/v) + fragments live without scratch spills.
#define SCR_PER_WAVE 2704   // floats: K 1024 | V 1024 | Q 512 | ATT 128 | A 16
__global__ void __launch_bounds__(128, 1) attn_kernel(
                            const float* __restrict__ feature, const float* __restrict__ xyz,
                            const float* __restrict__ bq, const float* __restrict__ bk,
                            const float* __restrict__ bv,
                            const unsigned short* __restrict__ wqb,   // contiguous wq|wk|wv bf16
                            const int* __restrict__ nidx, unsigned short* __restrict__ resb) {
  __shared__ unsigned short sWgt[3 * TDC * CPAD];   // 110,592 B: wq|wk|wv in LDS
  __shared__ float smem[4 * SCR_PER_WAVE];          // 43,264 B: per-wave attention scratch
  const int wave = threadIdx.x >> 5;
  const int lane = threadIdx.x & 31;
  const int half = lane >> 4;
  const int col  = lane & 15;

  // ---- TDM: DMA wq|wk|wv (110,592 B = 13,824 x 8B) from global into LDS (wave 0 only) ----
  if (threadIdx.x < 32) {
    unsigned long long ga = (unsigned long long)(uintptr_t)wqb;
    unsigned lds_off = (unsigned)(uintptr_t)(void*)sWgt;     // LDS byte offset (addr[31:0])
    const unsigned elems = (3u * TDC * CPAD * 2u) / 8u;      // 13824 (8-byte units)
    v4u g0;
    g0[0] = 1u;                                              // count=1, no gather, no restore
    g0[1] = lds_off;                                         // D#.lds_addr
    g0[2] = (unsigned)(ga & 0xFFFFFFFFu);                    // D#.global_addr[31:0]
    g0[3] = (unsigned)((ga >> 32) & 0x01FFFFFFu) | (2u << 30); // global_addr[56:32] | type=2
    v8i g1;
    g1[0] = (int)(3u << 16);                                 // workgroup_mask=0, data_size=3 (8B)
    g1[1] = (int)((elems & 0xFFFFu) << 16);                  // tensor_dim0[15:0]
    g1[2] = (int)(((elems >> 16) & 0xFFFFu) | (1u << 16));   // tensor_dim0[31:16], tensor_dim1=1
    g1[3] = (int)((elems & 0xFFFFu) << 16);                  // tensor_dim1[31:16]=0, tile_dim0=elems
    g1[4] = 1;                                               // tile_dim1=1, tile_dim2=0
    g1[5] = (int)elems;                                      // tensor_dim0_stride[31:0]
    g1[6] = 0;                                               // stride hi / dim1_stride lo
    g1[7] = 0;                                               // dim1_stride hi
    v4i g2 = {0, 0, 0, 0}, g3 = {0, 0, 0, 0};
    v8i g4 = {0, 0, 0, 0, 0, 0, 0, 0};                       // unused trailing group (clang-23 form)
    __builtin_amdgcn_tensor_load_to_lds(g0, g1, g2, g3, g4, 0);
    __builtin_amdgcn_s_wait_tensorcnt(0);                    // s_wait_tensorcnt 0
  }
  __syncthreads();                                           // publish LDS weights to all waves

  const unsigned short* wq_l = sWgt;
  const unsigned short* wk_l = sWgt + TDC * CPAD;
  const unsigned short* wv_l = sWgt + 2 * TDC * CPAD;

  float* sw   = smem + wave * SCR_PER_WAVE;
  float* sK   = sw;            // [64][16]
  float* sV   = sw + 1024;     // [64][16]
  float* sQ   = sw + 2048;     // [64][8]
  float* sATT = sw + 2560;     // [8][16]
  float* sA   = sw + 2688;     // [16]

  int p = blockIdx.x * 4 + wave;
  int b = p / NN, f = p % NN;
  const float* fb = feature + (size_t)b * CC * NN;
  const float* xb = xyz + (size_t)b * 3 * NN;
  int nbr = nidx[((size_t)b * NN + f) * K1C + col];  // this lane's neighbor column

  v8f zero = {0.f, 0.f, 0.f, 0.f, 0.f, 0.f, 0.f, 0.f};
  v8f accQ[4], accK[4], accV[4];
#pragma unroll
  for (int t = 0; t < 4; ++t) { accQ[t] = zero; accK[t] = zero; accV[t] = zero; }

  for (int kk = 0; kk < KSTEPS; ++kk) {
    // B-fragment of rel_pos (32 x 16): lane<16 -> col, K=kk*32+0..15; lane>=16 -> K=+16..31
    Frag16 R;
    int rb = kk * 32 + half * 16;
#pragma unroll
    for (int e = 0; e < 16; ++e) {
      int r = rb + e;
      float val;
      if      (r < 3)    val = xb[r * NN + nbr] - xb[r * NN + f];        // rel_xyz
      else if (r < 6)    val = xb[(r - 3) * NN + f];                     // xyz broadcast
      else if (r < 134)  val = fb[(r - 6) * NN + nbr] - fb[(r - 6) * NN + f]; // rel_fea
      else if (r < CINC) val = fb[(r - 134) * NN + f];                   // feature broadcast
      else               val = 0.f;                                      // K padding
      R.s[e] = f2bf(val);
    }
    // A-fragments (16x32) from LDS: lane<16 -> K 0..15; lane>=16 -> K 16..31
    int kbA = kk * 32 + half * 16;
#pragma unroll
    for (int t = 0; t < 4; ++t) {
      const unsigned short* aq = wq_l + (t * 16 + col) * CPAD + kbA;
      const unsigned short* ak = wk_l + (t * 16 + col) * CPAD + kbA;
      const unsigned short* av = wv_l + (t * 16 + col) * CPAD + kbA;
      Frag16 Aq, Ak, Av;
      Aq.q[0] = *(const uint4*)aq;  Aq.q[1] = *(const uint4*)(aq + 8);
      Ak.q[0] = *(const uint4*)ak;  Ak.q[1] = *(const uint4*)(ak + 8);
      Av.q[0] = *(const uint4*)av;  Av.q[1] = *(const uint4*)(av + 8);
      accQ[t] = __builtin_amdgcn_wmma_f32_16x16x32_bf16(false, Aq.v, false, R.v, (short)0, accQ[t], false, false);
      accK[t] = __builtin_amdgcn_wmma_f32_16x16x32_bf16(false, Ak.v, false, R.v, (short)0, accK[t], false, false);
      accV[t] = __builtin_amdgcn_wmma_f32_16x16x32_bf16(false, Av.v, false, R.v, (short)0, accV[t], false, false);
    }
  }

  // bias + stage to per-wave LDS (D layout: lane<16 -> M=r, lane>=16 -> M=8+r)
#pragma unroll
  for (int t = 0; t < 4; ++t)
#pragma unroll
    for (int r = 0; r < 8; ++r) {
      int m = t * 16 + r + half * 8;
      sK[m * 16 + col] = accK[t][r] + bk[m];
      sV[m * 16 + col] = accV[t][r] + bv[m];
      if (col < 8) sQ[m * 8 + col] = accQ[t][r] + bq[m];  // q valid for first 8 neighbors
    }
  __syncthreads();

  // attn[n][j] = sum_m q[m][n] * k[m][j]  (128 pairs / 32 lanes)
#pragma unroll
  for (int i = 0; i < 4; ++i) {
    int pp = i * 32 + lane;
    int n = pp >> 4, j = pp & 15;
    float a = 0.f;
    for (int m = 0; m < TDC; ++m) a += sQ[m * 8 + n] * sK[m * 16 + j];
    sATT[n * 16 + j] = a;
  }
  __syncthreads();
  if (lane < 8) {                               // softmax over j per n
    int n = lane;
    float mx = -3.4e38f;
    for (int j = 0; j < K1C; ++j) mx = fmaxf(mx, sATT[n * 16 + j]);
    float s = 0.f;
    for (int j = 0; j < K1C; ++j) { float e = __expf(sATT[n * 16 + j] - mx); sATT[n * 16 + j] = e; s += e; }
    float inv = 1.f / s;
    for (int j = 0; j < K1C; ++j) sATT[n * 16 + j] *= inv;
  }
  __syncthreads();
  if (lane < 16) {                              // collapse over n
    float a = 0.f;
    for (int n = 0; n < K2C; ++n) a += sATT[n * 16 + lane];
    sA[lane] = a;
  }
  __syncthreads();
#pragma unroll
  for (int h = 0; h < 2; ++h) {                 // res[m] = sum_j a[j] * v[m][j]
    int m = h * 32 + lane;
    float r = 0.f;
#pragma unroll
    for (int j = 0; j < K1C; ++j) r += sA[j] * sV[m * 16 + j];
    resb[(size_t)p * TDC + m] = f2bf(r);        // column-major bf16 [point][64]
  }
}

// ---------------- kernel 4: out = w1 @ res + b1 + feature (WMMA GEMM) ----------------
__global__ void __launch_bounds__(128) out_kernel(
                           const float* __restrict__ feature, const float* __restrict__ b1,
                           const unsigned short* __restrict__ w1b,
                           const unsigned short* __restrict__ resb,
                           float* __restrict__ out) {
  const int wave = threadIdx.x >> 5;
  const int lane = threadIdx.x & 31;
  const int half = lane >> 4;
  const int cl   = lane & 15;
  int col0 = blockIdx.x * 16;                  // 16 points per block (same batch: N % 16 == 0)
  int b = col0 / NN, f0 = col0 % NN;

  v8f acc[2];
  v8f zero = {0.f, 0.f, 0.f, 0.f, 0.f, 0.f, 0.f, 0.f};
  acc[0] = zero; acc[1] = zero;
#pragma unroll
  for (int kk = 0; kk < 2; ++kk) {
    Frag16 Bf;                                 // B 32x16: contiguous 16 K per lane
    const unsigned short* bp = resb + (size_t)(col0 + cl) * TDC + kk * 32 + half * 16;
    Bf.q[0] = *(const uint4*)bp; Bf.q[1] = *(const uint4*)(bp + 8);
    int kbA = kk * 32 + half * 16;
#pragma unroll
    for (int t = 0; t < 2; ++t) {
      int mt = wave * 2 + t;                   // 8 M-tiles over C=128, 2 per wave
      const unsigned short* ap = w1b + (mt * 16 + cl) * TDC + kbA;
      Frag16 Af;
      Af.q[0] = *(const uint4*)ap; Af.q[1] = *(const uint4*)(ap + 8);
      acc[t] = __builtin_amdgcn_wmma_f32_16x16x32_bf16(false, Af.v, false, Bf.v, (short)0, acc[t], false, false);
    }
  }
#pragma unroll
  for (int t = 0; t < 2; ++t) {
    int mt = wave * 2 + t;
#pragma unroll
    for (int r = 0; r < 8; ++r) {
      int c = mt * 16 + r + half * 8;
      size_t o = ((size_t)b * CC + c) * NN + (f0 + cl);
      out[o] = acc[t][r] + b1[c] + feature[o];  // + residual
    }
  }
}

// ---------------- host launcher ----------------
extern "C" void kernel_launch(void* const* d_in, const int* in_sizes, int n_in,
                              void* d_out, int out_size, void* d_ws, size_t ws_size,
                              hipStream_t stream) {
  const float* feature = (const float*)d_in[0];
  const float* xyz     = (const float*)d_in[1];
  const float* wq      = (const float*)d_in[2];
  const float* bq      = (const float*)d_in[3];
  const float* wk      = (const float*)d_in[4];
  const float* bk      = (const float*)d_in[5];
  const float* wv      = (const float*)d_in[6];
  const float* bv      = (const float*)d_in[7];
  const float* w1      = (const float*)d_in[8];
  const float* b1      = (const float*)d_in[9];
  float* out = (float*)d_out;

  // workspace layout (~3.2 MB)
  char* ws = (char*)d_ws;
  unsigned short* wqb = (unsigned short*)ws;                 // [64][288] bf16 (wq|wk|wv contiguous)
  unsigned short* wkb = wqb + TDC * CPAD;
  unsigned short* wvb = wkb + TDC * CPAD;
  unsigned short* w1b = wvb + TDC * CPAD;                    // [128][64] bf16
  size_t wbytes = (size_t)(3 * TDC * CPAD + CC * TDC) * 2;   // 126976 (128B aligned)
  int* nidx = (int*)(ws + wbytes);                           // [B][N][16] int
  unsigned short* resb = (unsigned short*)(ws + wbytes + (size_t)BB * NN * K1C * 4); // [B*N][64] bf16

  int prep_total = 3 * TDC * CPAD + CC * TDC;
  prep_weights<<<(prep_total + 255) / 256, 256, 0, stream>>>(wq, wk, wv, w1, wqb, wkb, wvb, w1b);
  knn_kernel<<<BB * (NN / 256), 256, 0, stream>>>(xyz, nidx);
  attn_kernel<<<(BB * NN) / 4, 128, 0, stream>>>(feature, xyz, bq, bk, bv, wqb, nidx, resb);
  out_kernel<<<(BB * NN) / 16, 128, 0, stream>>>(feature, b1, w1b, resb, out);
}
